// EncoderLayer_6622839571235
// MI455X (gfx1250) — matrix operand order repaired and checked by reading, not compiled, not run on hardware
//
#include <hip/hip_runtime.h>

// B=4, S=2048, D=1024, H=16, DK=64, DFF=4096, M=B*S=8192
#define BB   4
#define SS   2048
#define DD   1024
#define HH   16
#define DKK  64
#define DFFN 4096
#define MM   8192

typedef _Float16 v16h __attribute__((ext_vector_type(16)));
typedef _Float16 v8h  __attribute__((ext_vector_type(8)));
typedef float    v8f  __attribute__((ext_vector_type(8)));
typedef int      v4i  __attribute__((ext_vector_type(4)));

// gfx1250 async global->LDS path (guarded: host pass / older toolchains fall back)
#if __has_builtin(__builtin_amdgcn_global_load_async_to_lds_b128) && \
    __has_builtin(__builtin_amdgcn_s_wait_asynccnt)
#define USE_ASYNC 1
#else
#define USE_ASYNC 0
#endif

#if USE_ASYNC
// 16-byte async copy: global -> LDS, immediate offset applies to both sides.
template <int OFF>
__device__ __forceinline__ void async_cp16(const _Float16* g, _Float16* l) {
  __builtin_amdgcn_global_load_async_to_lds_b128(
      (__attribute__((address_space(1))) v4i*)(uintptr_t)g,
      (__attribute__((address_space(3))) v4i*)(uintptr_t)(unsigned)(uintptr_t)l,
      OFF, 0);
}
#endif

__device__ __forceinline__ v16h cat8(v8h lo, v8h hi) {
  return __builtin_shufflevector(lo, hi, 0,1,2,3,4,5,6,7,8,9,10,11,12,13,14,15);
}

__device__ __forceinline__ v8f wmma_f16(v16h a, v16h b, v8f c) {
  // D = A(16x32 f16) * B(32x16 f16) + C(16x16 f32)
  return __builtin_amdgcn_wmma_f32_16x16x32_f16(false, a, false, b, (short)0, c, false, false);
}

// ---------------------------------------------------------------------------
// fp32 [K][N] -> f16 [N][K]  (transposed so GEMM B-fragments are contiguous-K)
// ---------------------------------------------------------------------------
__global__ __launch_bounds__(256) void convertT(const float* __restrict__ w,
                                                _Float16* __restrict__ wT,
                                                int K, int N) {
  size_t idx = (size_t)blockIdx.x * 256 + threadIdx.x;
  int k = (int)(idx / (size_t)N);
  int n = (int)(idx % (size_t)N);
  wT[(size_t)n * K + k] = (_Float16)w[idx];
}

// ---------------------------------------------------------------------------
// LayerNorm over D=1024, fp32 in -> f16 out. 256 threads/row, 4 elems/thread.
// ---------------------------------------------------------------------------
__global__ __launch_bounds__(256) void layernorm_f16(const float* __restrict__ x,
                                                     const float* __restrict__ g,
                                                     const float* __restrict__ be,
                                                     _Float16* __restrict__ out) {
  const int row = blockIdx.x;
  const int tid = threadIdx.x;
  const float* xr = x + (size_t)row * DD;
  float4 v = ((const float4*)xr)[tid];
  float s  = v.x + v.y + v.z + v.w;
  float sq = v.x*v.x + v.y*v.y + v.z*v.z + v.w*v.w;
  #pragma unroll
  for (int off = 1; off < 32; off <<= 1) {
    s  += __shfl_xor(s,  off);
    sq += __shfl_xor(sq, off);
  }
  __shared__ float ssum[8], ssq[8];
  if ((tid & 31) == 0) { ssum[tid >> 5] = s; ssq[tid >> 5] = sq; }
  __syncthreads();
  float st = 0.f, sqt = 0.f;
  #pragma unroll
  for (int i = 0; i < 8; ++i) { st += ssum[i]; sqt += ssq[i]; }
  float mu  = st * (1.0f / DD);
  float var = sqt * (1.0f / DD) - mu * mu;
  float rs  = rsqrtf(var + 1e-5f);
  int c = tid * 4;
  float4 gg = *(const float4*)(g + c);
  float4 bb = *(const float4*)(be + c);
  _Float16* o = out + (size_t)row * DD + c;
  o[0] = (_Float16)((v.x - mu) * rs * gg.x + bb.x);
  o[1] = (_Float16)((v.y - mu) * rs * gg.y + bb.y);
  o[2] = (_Float16)((v.z - mu) * rs * gg.z + bb.z);
  o[3] = (_Float16)((v.w - mu) * rs * gg.w + bb.w);
}

// ---------------------------------------------------------------------------
// WMMA GEMM: C[M,N] = A16[M,K] * BT16[N,K]^T  (+bias)(+relu)(+residual)
// Block tile 128x64, K-step 64, 8 waves; wave w owns rows [w*16, w*16+16).
// A/B tiles staged to LDS with gfx1250 async-to-LDS engine (ASYNCcnt).
// MODE 0: f16 store [M][N]   MODE 1: f16 store transposed per-head [B][H][DK][S]
// MODE 2: fp32 store out = res + acc (+bias)
// ---------------------------------------------------------------------------
template <int MODE, bool RELU>
__global__ __launch_bounds__(256) void gemm_ws(const _Float16* __restrict__ A,
                                               const _Float16* __restrict__ BT,
                                               const float* __restrict__ bias,
                                               const float* __restrict__ res,
                                               float* __restrict__ outF,
                                               _Float16* __restrict__ outH,
                                               int Mdim, int Ndim, int Kdim) {
  __shared__ _Float16 lA[128 * 80];  // 20 KB, +16-half row pad
  __shared__ _Float16 lB[64 * 80];   // 10 KB
  const int tid  = threadIdx.x;
  const int lane = tid & 31, wave = tid >> 5;
  const int m0 = blockIdx.y * 128;
  const int n0 = blockIdx.x * 64;
  const int m  = lane & 15, hi = lane >> 4;

  const int ra = tid >> 1, ca = (tid & 1) * 32;   // A stage: 64 B / thread
  const int rb = tid >> 2, cb = (tid & 3) * 16;   // B stage: 32 B / thread

  v8f acc[4] = {};

  for (int kt = 0; kt < Kdim; kt += 64) {
    const _Float16* ga = A  + (size_t)(m0 + ra) * Kdim + kt + ca;
    const _Float16* gb = BT + (size_t)(n0 + rb) * Kdim + kt + cb;
    _Float16* la = &lA[ra * 80 + ca];
    _Float16* lb = &lB[rb * 80 + cb];
#if USE_ASYNC
    async_cp16<0>(ga, la);
    async_cp16<16>(ga, la);
    async_cp16<32>(ga, la);
    async_cp16<48>(ga, la);
    async_cp16<0>(gb, lb);
    async_cp16<16>(gb, lb);
#else
    ((v16h*)la)[0] = ((const v16h*)ga)[0];
    *(v16h*)(la + 16) = ((const v16h*)ga)[1];
    *(v16h*)lb = *(const v16h*)gb;
#endif
    if (kt + 64 < Kdim) {  // prefetch next K tile into L2/WGP$
      __builtin_prefetch(ga + 64, 0, 1);
      __builtin_prefetch(gb + 64, 0, 1);
    }
#if USE_ASYNC
    __builtin_amdgcn_s_wait_asynccnt(0);
#endif
    __syncthreads();
    #pragma unroll
    for (int ks = 0; ks < 2; ++ks) {
      const int kk = ks * 32;
      // A fragment, contiguous-K: lane half selects 16-wide K half (matches B)
      const _Float16* ap = &lA[(wave * 16 + m) * 80 + kk + hi * 16];
      v16h a = cat8(*(const v8h*)ap, *(const v8h*)(ap + 8));
      #pragma unroll
      for (int nt = 0; nt < 4; ++nt) {
        v16h bf = *(const v16h*)&lB[(nt * 16 + m) * 80 + kk + hi * 16];
        acc[nt] = wmma_f16(a, bf, acc[nt]);
      }
    }
    __syncthreads();
  }

  // epilogue — C/D layout: lane half holds rows r / r+8, col = lane&15
  #pragma unroll
  for (int nt = 0; nt < 4; ++nt) {
    const int col = n0 + nt * 16 + m;
    const float bv = bias ? bias[col] : 0.0f;
    #pragma unroll
    for (int r = 0; r < 8; ++r) {
      const int row = m0 + wave * 16 + (hi ? r + 8 : r);
      float v = acc[nt][r] + bv;
      if (RELU) v = fmaxf(v, 0.0f);
      if (MODE == 0) {
        outH[(size_t)row * Ndim + col] = (_Float16)v;
      } else if (MODE == 1) {
        const int b  = row >> 11;          // /S
        const int s  = row & (SS - 1);
        const int hh = col >> 6;           // /DK
        const int dk = col & (DKK - 1);
        outH[(((size_t)b * HH + hh) * DKK + dk) * SS + s] = (_Float16)v;
      } else {
        const size_t idx = (size_t)row * Ndim + col;
        outF[idx] = res[idx] + v;
      }
    }
  }
}

// ---------------------------------------------------------------------------
// Flash attention: one wave = 16 query rows, online softmax over key blocks
// of 32.  q16/k16: [B*S][D] f16; vT16: [B][H][DK][S] f16; out: [B*S][D] f16.
// ---------------------------------------------------------------------------
__global__ __launch_bounds__(256) void flash_attn(const _Float16* __restrict__ q16,
                                                  const _Float16* __restrict__ k16,
                                                  const _Float16* __restrict__ vT16,
                                                  _Float16* __restrict__ attn16) {
  __shared__ _Float16 ldsP[8 * 16 * 40];  // per-wave 16x32 (+8 pad) P scratch
  const int tid  = threadIdx.x;
  const int lane = tid & 31, wave = tid >> 5;
  const int m = lane & 15, hi = lane >> 4;
  const int b = blockIdx.z, h = blockIdx.y;
  const int q0 = blockIdx.x * 128 + wave * 16;

  // Q fragments (DK=64 -> two 16x32 A-frags)
  const _Float16* qrow = q16 + ((size_t)(b * SS + q0 + m)) * DD + h * DKK;
  v16h aq[2];
  #pragma unroll
  for (int ks = 0; ks < 2; ++ks) {
    const _Float16* p = qrow + ks * 32 + hi * 16;
    aq[ks] = cat8(*(const v8h*)p, *(const v8h*)(p + 8));
  }

  float mi[8], li[8];
  #pragma unroll
  for (int r = 0; r < 8; ++r) { mi[r] = -1e30f; li[r] = 0.0f; }
  v8f acc[4] = {};
  _Float16* lp = &ldsP[wave * 16 * 40];

  for (int kb = 0; kb < SS; kb += 32) {
    if (kb + 32 < SS) {  // prefetch next key/value block
      __builtin_prefetch(k16 + ((size_t)(b * SS + kb + 32 + m)) * DD + h * DKK, 0, 1);
      __builtin_prefetch(vT16 + (((size_t)(b * HH + h)) * DKK + m) * SS + kb + 32, 0, 1);
    }
    v8f s0 = {}, s1 = {};
    #pragma unroll
    for (int ks = 0; ks < 2; ++ks) {
      const int kk = ks * 32;
      const _Float16* kbase =
          k16 + ((size_t)(b * SS + kb + m)) * DD + h * DKK + kk + hi * 16;
      v16h bk0 = *(const v16h*)kbase;
      v16h bk1 = *(const v16h*)(kbase + 16 * DD);
      s0 = wmma_f16(aq[ks], bk0, s0);
      s1 = wmma_f16(aq[ks], bk1, s1);
    }
    #pragma unroll
    for (int r = 0; r < 8; ++r) {
      float x0 = s0[r] * 0.125f;   // 1/sqrt(DK)
      float x1 = s1[r] * 0.125f;
      float mx = fmaxf(x0, x1);
      mx = fmaxf(mx, __shfl_xor(mx, 1));
      mx = fmaxf(mx, __shfl_xor(mx, 2));
      mx = fmaxf(mx, __shfl_xor(mx, 4));
      mx = fmaxf(mx, __shfl_xor(mx, 8));
      float mnew = fmaxf(mi[r], mx);
      float p0 = __expf(x0 - mnew);
      float p1 = __expf(x1 - mnew);
      float rs = p0 + p1;
      rs += __shfl_xor(rs, 1);
      rs += __shfl_xor(rs, 2);
      rs += __shfl_xor(rs, 4);
      rs += __shfl_xor(rs, 8);
      float corr = __expf(mi[r] - mnew);
      li[r] = li[r] * corr + rs;
      mi[r] = mnew;
      #pragma unroll
      for (int nt = 0; nt < 4; ++nt) acc[nt][r] *= corr;
      // C-layout -> row-major P in wave-private LDS (wave DS ops are in-order)
      const int prow = hi ? r + 8 : r;
      lp[prow * 40 + m]      = (_Float16)p0;
      lp[prow * 40 + 16 + m] = (_Float16)p1;
    }
    // reload P as an A-fragment (16x32)
    const _Float16* pr = lp + m * 40 + hi * 16;
    v16h ap = cat8(*(const v8h*)pr, *(const v8h*)(pr + 8));
    #pragma unroll
    for (int nt = 0; nt < 4; ++nt) {
      const _Float16* vb =
          vT16 + (((size_t)(b * HH + h)) * DKK + nt * 16 + m) * SS + kb + hi * 16;
      acc[nt] = wmma_f16(ap, *(const v16h*)vb, acc[nt]);
    }
  }

  #pragma unroll
  for (int r = 0; r < 8; ++r) {
    const float inv = li[r] > 0.0f ? 1.0f / li[r] : 0.0f;  // nan_to_num guard
    const int row = q0 + (hi ? r + 8 : r);
    #pragma unroll
    for (int nt = 0; nt < 4; ++nt) {
      attn16[((size_t)(b * SS + row)) * DD + h * DKK + nt * 16 + m] =
          (_Float16)(acc[nt][r] * inv);
    }
  }
}

// ---------------------------------------------------------------------------
extern "C" void kernel_launch(void* const* d_in, const int* in_sizes, int n_in,
                              void* d_out, int out_size, void* d_ws, size_t ws_size,
                              hipStream_t stream) {
  const float* x   = (const float*)d_in[0];
  const float* wq  = (const float*)d_in[1];
  const float* wk  = (const float*)d_in[2];
  const float* wv  = (const float*)d_in[3];
  const float* wo  = (const float*)d_in[4];
  const float* w1  = (const float*)d_in[5];
  const float* b1  = (const float*)d_in[6];
  const float* w2  = (const float*)d_in[7];
  const float* b2  = (const float*)d_in[8];
  const float* g1  = (const float*)d_in[9];
  const float* be1 = (const float*)d_in[10];
  const float* g2  = (const float*)d_in[11];
  const float* be2 = (const float*)d_in[12];
  float* out = (float*)d_out;

  char* ws = (char*)d_ws;
  const size_t MB = 1ull << 20;
  _Float16* wqT   = (_Float16*)(ws + 0 * MB);    // 2 MB
  _Float16* wkT   = (_Float16*)(ws + 2 * MB);    // 2 MB
  _Float16* wvT   = (_Float16*)(ws + 4 * MB);    // 2 MB
  _Float16* woT   = (_Float16*)(ws + 6 * MB);    // 2 MB
  _Float16* w1T   = (_Float16*)(ws + 8 * MB);    // 8 MB
  _Float16* w2T   = (_Float16*)(ws + 16 * MB);   // 8 MB
  _Float16* h16   = (_Float16*)(ws + 24 * MB);   // 16 MB (reused as h2)
  _Float16* q16   = (_Float16*)(ws + 40 * MB);   // 16 MB
  _Float16* k16   = (_Float16*)(ws + 56 * MB);   // 16 MB
  _Float16* vT16  = (_Float16*)(ws + 72 * MB);   // 16 MB
  _Float16* att16 = (_Float16*)(ws + 88 * MB);   // 16 MB
  _Float16* h2_16 = h16;                          // alias: h16 dead after QKV
  _Float16* ffn16 = q16;                          // alias: q/k/vT/att dead (64 MB)

  // 1) weight convert+transpose to f16 [N][K]
  convertT<<<(DD * DD) / 256, 256, 0, stream>>>(wq, wqT, DD, DD);
  convertT<<<(DD * DD) / 256, 256, 0, stream>>>(wk, wkT, DD, DD);
  convertT<<<(DD * DD) / 256, 256, 0, stream>>>(wv, wvT, DD, DD);
  convertT<<<(DD * DD) / 256, 256, 0, stream>>>(wo, woT, DD, DD);
  convertT<<<(DD * DFFN) / 256, 256, 0, stream>>>(w1, w1T, DD, DFFN);
  convertT<<<(DFFN * DD) / 256, 256, 0, stream>>>(w2, w2T, DFFN, DD);

  // 2) LN1
  layernorm_f16<<<MM, 256, 0, stream>>>(x, g1, be1, h16);

  // 3) Q,K,V projections
  dim3 gD(DD / 64, MM / 128);       // 16 x 64
  gemm_ws<0, false><<<gD, 256, 0, stream>>>(h16, wqT, nullptr, nullptr, nullptr, q16,  MM, DD, DD);
  gemm_ws<0, false><<<gD, 256, 0, stream>>>(h16, wkT, nullptr, nullptr, nullptr, k16,  MM, DD, DD);
  gemm_ws<1, false><<<gD, 256, 0, stream>>>(h16, wvT, nullptr, nullptr, nullptr, vT16, MM, DD, DD);

  // 4) attention
  dim3 gA(SS / 128, HH, BB);        // 16 x 16 x 4
  flash_attn<<<gA, 256, 0, stream>>>(q16, k16, vT16, att16);

  // 5) out-proj + residual -> d_out (fp32)
  gemm_ws<2, false><<<gD, 256, 0, stream>>>(att16, woT, nullptr, x, out, nullptr, MM, DD, DD);

  // 6) LN2
  layernorm_f16<<<MM, 256, 0, stream>>>(out, g2, be2, h2_16);

  // 7) FFN1: relu(h2 @ w1 + b1) -> f16
  dim3 gF(DFFN / 64, MM / 128);     // 64 x 64
  gemm_ws<0, true><<<gF, 256, 0, stream>>>(h2_16, w1T, b1, nullptr, nullptr, ffn16, MM, DFFN, DD);

  // 8) FFN2 + bias + residual (in place on d_out)
  gemm_ws<2, false><<<gD, 256, 0, stream>>>(ffn16, w2T, b2, out, out, nullptr, MM, DD, DFFN);
}
